// TemporalAttentionLayer_36223754175173
// MI455X (gfx1250) — matrix operand order repaired and checked by reading, not compiled, not run on hardware
//
#include <hip/hip_runtime.h>
#include <cstdint>
#include <cstddef>

// ---------------------------------------------------------------------------
// Types
// ---------------------------------------------------------------------------
typedef __bf16 bf16;
typedef __attribute__((ext_vector_type(16))) __bf16 v16bf;
typedef __attribute__((ext_vector_type(8)))  float  v8f;

#define T_DIM 32
#define B_DIM 32
#define J_DIM 42
#define D_DIM 256
#define H_DIM 8
#define HD_DIM 32
#define FF_DIM 2048
#define ROWS (T_DIM*B_DIM*J_DIM)      // 43008
#define TB (T_DIM*B_DIM)              // 1024
#define BJ (B_DIM*J_DIM)              // 1344
#define EPS 1e-5f

__device__ inline bf16 f2bf(float f) {
    union { float f; uint32_t u; } c; c.f = f;
    uint32_t u = c.u;
    uint32_t r = u + 0x7FFFu + ((u >> 16) & 1u);
    union { unsigned short s; bf16 b; } o; o.s = (unsigned short)(r >> 16);
    return o.b;
}
__device__ inline float bf2f(bf16 b) {
    union { unsigned short s; bf16 b; } i; i.b = b;
    union { uint32_t u; float f; } o; o.u = ((uint32_t)i.s) << 16;
    return o.f;
}
__device__ inline float wred_sum(float v) {
    #pragma unroll
    for (int o = 16; o; o >>= 1) v += __shfl_xor(v, o, 32);
    return v;
}
__device__ inline float wred_max(float v) {
    #pragma unroll
    for (int o = 16; o; o >>= 1) v = fmaxf(v, __shfl_xor(v, o, 32));
    return v;
}

// gfx1250 async global->LDS copy (ASYNCcnt-tracked, per-lane 16B).
// Flat pointer to __shared__ has the LDS byte offset in its low 32 bits.
__device__ inline uint32_t lds_off32(const void* p) {
    return (uint32_t)(uintptr_t)p;
}
__device__ inline void async_ld_b128(uint32_t lds_byte_off, const void* gaddr) {
    asm volatile("global_load_async_to_lds_b128 %0, %1, off"
                 :: "v"(lds_byte_off), "v"(gaddr) : "memory");
}
__device__ inline void wait_async0() {
    asm volatile("s_wait_asynccnt 0x0" ::: "memory");
}

// ---------------------------------------------------------------------------
// Weight f32 -> bf16 conversion
// ---------------------------------------------------------------------------
__global__ void k_f32_to_bf16(const float* __restrict__ src, bf16* __restrict__ dst, int n) {
    int i = blockIdx.x * 256 + threadIdx.x;
    if (i < n) dst[i] = f2bf(src[i]);
}

// ---------------------------------------------------------------------------
// Pack src [t,b,j,d] -> xp [(j*TB + tb), d]   (stage-1 row order)
// ---------------------------------------------------------------------------
__global__ void k_pack_x1(const float* __restrict__ src, float* __restrict__ xp) {
    int q = blockIdx.x;           // 0..ROWS-1 in r1 order
    int d = threadIdx.x;
    int j = q >> 10;              // /1024
    int tb = q & 1023;
    xp[(size_t)q * D_DIM + d] = src[((size_t)tb * J_DIM + j) * D_DIM + d];
}

// ---------------------------------------------------------------------------
// Tiled WMMA GEMM:  C[M,N] = A[M,K] @ W[N,K]^T (+bias) (+relu)
//   BM=128 BN=128 BK=32, 256 threads = 8 waves (4x2), each wave 32x64 output
//   bf16 tiles staged with global_load_async_to_lds_b128 where possible.
// ---------------------------------------------------------------------------
#define BM 128
#define BN 128
#define BK 32
#define LDS_STRIDE 40   // bf16 elems per row -> 80B, 16B aligned

template<bool ABF, bool OBF, bool RELU, bool HASBIAS>
__global__ __launch_bounds__(256) void k_gemm_wmma(
    const void* __restrict__ Av, const bf16* __restrict__ Bw,
    const float* __restrict__ bias, void* __restrict__ Cv,
    int M, int N, int K)
{
    __shared__ __align__(16) bf16 As[BM * LDS_STRIDE];
    __shared__ __align__(16) bf16 Bs[BN * LDS_STRIDE];

    const int tid  = threadIdx.x;
    const int wave = tid >> 5, lane = tid & 31;
    const int half = lane >> 4, lr = lane & 15;
    const int wm = wave >> 1, wn = wave & 1;   // 4 x 2 waves
    const int mBase = blockIdx.y * BM;
    const int nBase = blockIdx.x * BN;

    const float* Af = (const float*)Av;
    const bf16*  Ab = (const bf16*)Av;

    // staging mapping: thread covers one 128x32 row-half (16 elems)
    const int sr  = tid >> 1;            // 0..127
    const int sc0 = (tid & 1) * 16;      // 0 or 16 (elements)

    v8f acc[2][4];
    #pragma unroll
    for (int i = 0; i < 2; i++)
        #pragma unroll
        for (int j = 0; j < 4; j++)
            #pragma unroll
            for (int r = 0; r < 8; r++) acc[i][j][r] = 0.f;

    // hoist bias (per-lane column) into registers
    float bv[4];
    #pragma unroll
    for (int tn = 0; tn < 4; tn++)
        bv[tn] = HASBIAS ? bias[nBase + wn * 64 + tn * 16 + lr] : 0.f;

    for (int k0 = 0; k0 < BK * ((K + BK - 1) / BK); k0 += BK) {
        // prefetch next K-tile (global_prefetch_b8)
        if (k0 + BK < K) {
            __builtin_prefetch(Bw + (size_t)(nBase + sr) * K + k0 + BK, 0, 1);
            if (ABF) __builtin_prefetch(Ab + (size_t)(mBase + sr) * K + k0 + BK, 0, 1);
            else     __builtin_prefetch(Af + (size_t)(mBase + sr) * K + k0 + BK, 0, 1);
        }
        // ---- stage A tile (128x32) ----
        if (ABF) {
            const bf16* g = Ab + (size_t)(mBase + sr) * K + k0 + sc0;
            uint32_t l = lds_off32(&As[sr * LDS_STRIDE + sc0]);
            async_ld_b128(l,      g);
            async_ld_b128(l + 16, g + 8);
        } else {
            const float4* g = (const float4*)(Af + (size_t)(mBase + sr) * K + k0 + sc0);
            float4 f0 = g[0], f1 = g[1], f2 = g[2], f3 = g[3];
            union { uint4 q; bf16 h[8]; } p0, p1;
            p0.h[0] = f2bf(f0.x); p0.h[1] = f2bf(f0.y); p0.h[2] = f2bf(f0.z); p0.h[3] = f2bf(f0.w);
            p0.h[4] = f2bf(f1.x); p0.h[5] = f2bf(f1.y); p0.h[6] = f2bf(f1.z); p0.h[7] = f2bf(f1.w);
            p1.h[0] = f2bf(f2.x); p1.h[1] = f2bf(f2.y); p1.h[2] = f2bf(f2.z); p1.h[3] = f2bf(f2.w);
            p1.h[4] = f2bf(f3.x); p1.h[5] = f2bf(f3.y); p1.h[6] = f2bf(f3.z); p1.h[7] = f2bf(f3.w);
            *(uint4*)(As + sr * LDS_STRIDE + sc0)     = p0.q;
            *(uint4*)(As + sr * LDS_STRIDE + sc0 + 8) = p1.q;
        }
        // ---- stage B tile (128n x 32k) via async copy ----
        {
            const bf16* g = Bw + (size_t)(nBase + sr) * K + k0 + sc0;
            uint32_t l = lds_off32(&Bs[sr * LDS_STRIDE + sc0]);
            async_ld_b128(l,      g);
            async_ld_b128(l + 16, g + 8);
        }
        wait_async0();
        __syncthreads();

        // ---- fragments (ds_load_b128) per documented CDNA5 layouts ----
        union FU { v16bf v; uint4 q[2]; };
        FU afrag[2], bfrag[4];
        #pragma unroll
        for (int tm = 0; tm < 2; tm++) {
            const bf16* ap = &As[(wm * 32 + tm * 16 + lr) * LDS_STRIDE + half * 8];
            afrag[tm].q[0] = *(const uint4*)(ap);
            afrag[tm].q[1] = *(const uint4*)(ap + 16);
        }
        #pragma unroll
        for (int tn = 0; tn < 4; tn++) {
            const bf16* bp = &Bs[(wn * 64 + tn * 16 + lr) * LDS_STRIDE + half * 8];
            bfrag[tn].q[0] = *(const uint4*)(bp);
            bfrag[tn].q[1] = *(const uint4*)(bp + 16);
        }
        #pragma unroll
        for (int tm = 0; tm < 2; tm++)
            #pragma unroll
            for (int tn = 0; tn < 4; tn++)
                acc[tm][tn] = __builtin_amdgcn_wmma_f32_16x16x32_bf16(
                    false, afrag[tm].v, false, bfrag[tn].v,
                    (short)0, acc[tm][tn], false, false);
        __syncthreads();
    }

    // ---- epilogue ----
    #pragma unroll
    for (int tm = 0; tm < 2; tm++) {
        #pragma unroll
        for (int tn = 0; tn < 4; tn++) {
            int col = nBase + wn * 64 + tn * 16 + lr;
            #pragma unroll
            for (int r = 0; r < 8; r++) {
                int row = mBase + wm * 32 + tm * 16 + half * 8 + r;
                float v = acc[tm][tn][r] + bv[tn];
                if (RELU) v = fmaxf(v, 0.f);
                if (OBF) ((bf16*)Cv)[(size_t)row * N + col] = f2bf(v);
                else     ((float*)Cv)[(size_t)row * N + col] = v;
            }
        }
    }
}

// ---------------------------------------------------------------------------
// Stage-1 attention: per (tb,h) problem, J=42, hd=32, softmax(QK^T/sqrt(32))V
// ---------------------------------------------------------------------------
__global__ __launch_bounds__(256) void k_attn1(const bf16* __restrict__ qkv,
                                               bf16* __restrict__ o1) {
    __shared__ float Qs[J_DIM][33], Ks[J_DIM][33], Vs[J_DIM][33], Ss[J_DIM][44];
    const int tid = threadIdx.x, wave = tid >> 5, lane = tid & 31;
    const int tb = blockIdx.x >> 3, h = blockIdx.x & 7;

    for (int idx = tid; idx < J_DIM * 32; idx += 256) {
        int j = idx >> 5, d = idx & 31;
        size_t base = ((size_t)(j * TB + tb)) * (3 * D_DIM) + h * HD_DIM + d;
        Qs[j][d] = bf2f(qkv[base]);
        Ks[j][d] = bf2f(qkv[base + D_DIM]);
        Vs[j][d] = bf2f(qkv[base + 2 * D_DIM]);
    }
    __syncthreads();

    for (int idx = tid; idx < J_DIM * J_DIM; idx += 256) {
        int q = idx / J_DIM, k = idx % J_DIM;
        float s = 0.f;
        #pragma unroll
        for (int d = 0; d < 32; d++) s += Qs[q][d] * Ks[k][d];
        Ss[q][k] = s * 0.17677669529663687f;   // 1/sqrt(32)
    }
    __syncthreads();

    for (int r = wave; r < J_DIM; r += 8) {
        float v0 = Ss[r][lane];
        float v1 = (lane < J_DIM - 32) ? Ss[r][32 + lane] : -3.0e38f;
        float m = wred_max(fmaxf(v0, v1));
        float e0 = expf(v0 - m);
        float e1 = (lane < J_DIM - 32) ? expf(v1 - m) : 0.f;
        float s = wred_sum(e0 + e1);
        float inv = 1.f / s;
        Ss[r][lane] = e0 * inv;
        if (lane < J_DIM - 32) Ss[r][32 + lane] = e1 * inv;
    }
    __syncthreads();

    for (int idx = tid; idx < J_DIM * 32; idx += 256) {
        int q = idx >> 5, d = idx & 31;
        float o = 0.f;
        for (int k = 0; k < J_DIM; k++) o += Ss[q][k] * Vs[k][d];
        o1[((size_t)(q * TB + tb)) * D_DIM + h * HD_DIM + d] = f2bf(o);
    }
}

// ---------------------------------------------------------------------------
// Stage-2 cosine attention: per (n=bj,h), T=32, L2-norm q/k, *scale + bias
// ---------------------------------------------------------------------------
__global__ __launch_bounds__(256) void k_attn2(const bf16* __restrict__ qkv,
                                               const float* __restrict__ pb,
                                               const float* __restrict__ scl,
                                               bf16* __restrict__ o2) {
    __shared__ float Qs[32][33], Ks[32][33], Vs[32][33], Ss[32][33];
    const int tid = threadIdx.x, wave = tid >> 5, lane = tid & 31;
    const int n = blockIdx.x >> 3, h = blockIdx.x & 7;

    for (int idx = tid; idx < 1024; idx += 256) {
        int t = idx >> 5, d = idx & 31;
        size_t base = ((size_t)(n * T_DIM + t)) * (3 * D_DIM) + h * HD_DIM + d;
        Qs[t][d] = bf2f(qkv[base]);
        Ks[t][d] = bf2f(qkv[base + D_DIM]);
        Vs[t][d] = bf2f(qkv[base + 2 * D_DIM]);
    }
    __syncthreads();

    for (int r = wave; r < 32; r += 8) {
        float q = Qs[r][lane];
        float sq = wred_sum(q * q);
        Qs[r][lane] = q / fmaxf(sqrtf(sq), 1e-12f);
        float k = Ks[r][lane];
        float sk = wred_sum(k * k);
        Ks[r][lane] = k / fmaxf(sqrtf(sk), 1e-12f);
    }
    __syncthreads();

    const float sc = scl[h];
    for (int idx = tid; idx < 1024; idx += 256) {
        int q = idx >> 5, k = idx & 31;
        float s = 0.f;
        #pragma unroll
        for (int d = 0; d < 32; d++) s += Qs[q][d] * Ks[k][d];
        Ss[q][k] = s * sc + pb[(q - k + T_DIM - 1) * H_DIM + h];
    }
    __syncthreads();

    for (int r = wave; r < 32; r += 8) {
        float v = Ss[r][lane];
        float m = wred_max(v);
        float e = expf(v - m);
        float s = wred_sum(e);
        Ss[r][lane] = e / s;
    }
    __syncthreads();

    for (int idx = tid; idx < 1024; idx += 256) {
        int q = idx >> 5, d = idx & 31;
        float o = 0.f;
        #pragma unroll
        for (int k = 0; k < 32; k++) o += Ss[q][k] * Vs[k][d];
        o2[((size_t)(n * T_DIM + q)) * D_DIM + h * HD_DIM + d] = f2bf(o);
    }
}

// ---------------------------------------------------------------------------
// Relative-position-bias MLP + logit scale (tiny, one block)
// ---------------------------------------------------------------------------
__global__ __launch_bounds__(256) void k_cpb(const float* __restrict__ w1,
                                             const float* __restrict__ b1,
                                             const float* __restrict__ lng,
                                             const float* __restrict__ lnb,
                                             const float* __restrict__ w2,
                                             const float* __restrict__ ls,
                                             float* __restrict__ pb,
                                             float* __restrict__ scl) {
    __shared__ float act[D_DIM];
    __shared__ float red[8];
    const int tid = threadIdx.x, wave = tid >> 5, lane = tid & 31;
    if (tid < H_DIM) scl[tid] = expf(fminf(ls[tid], 2.99573227355399f)); // ln(20)

    const float w1v = w1[tid], b1v = b1[tid], gv = lng[tid], bev = lnb[tid];
    for (int p = 0; p < 2 * T_DIM - 1; p++) {
        float pos = (float)(p - (T_DIM - 1));
        float hv = pos * w1v + b1v;
        float s = wred_sum(hv);
        if (lane == 0) red[wave] = s;
        __syncthreads();
        float mean = 0.f;
        #pragma unroll
        for (int i = 0; i < 8; i++) mean += red[i];
        mean *= (1.f / D_DIM);
        __syncthreads();
        float dlt = hv - mean;
        float s2 = wred_sum(dlt * dlt);
        if (lane == 0) red[wave] = s2;
        __syncthreads();
        float var = 0.f;
        #pragma unroll
        for (int i = 0; i < 8; i++) var += red[i];
        var *= (1.f / D_DIM);
        float xn = dlt * rsqrtf(var + EPS) * gv + bev;
        act[tid] = xn / (1.f + expf(-xn));      // silu
        __syncthreads();
        float a = 0.f;
        #pragma unroll
        for (int i = 0; i < 8; i++) {
            int d = lane + 32 * i;
            a += act[d] * w2[wave * D_DIM + d];
        }
        a = wred_sum(a);
        if (lane == 0) pb[p * H_DIM + wave] = a;
        __syncthreads();
    }
}

// ---------------------------------------------------------------------------
// Residual + LayerNorm, wave-per-row, with inter-stage row permutation
// ---------------------------------------------------------------------------
__global__ __launch_bounds__(256) void k_ln(const float* __restrict__ A,
                                            const float* __restrict__ Bm,
                                            const float* __restrict__ g,
                                            const float* __restrict__ be,
                                            float* __restrict__ out,
                                            int mode, int nrows) {
    const int wave = threadIdx.x >> 5, lane = threadIdx.x & 31;
    const int q = blockIdx.x * 8 + wave;
    if (q >= nrows) return;
    int rOut = q;
    if (mode == 0) {
        int j = q >> 10, tb = q & 1023;
        int t = tb >> 5, b = tb & 31;
        rOut = (b * J_DIM + j) * T_DIM + t;
    } else if (mode == 1) {
        int n = q >> 5, t = q & 31;
        int b = n / J_DIM, j = n % J_DIM;
        rOut = (t * B_DIM + b) * J_DIM + j;
    }
    float x[8];
    size_t ib = (size_t)q * D_DIM;
    float s = 0.f;
    #pragma unroll
    for (int i = 0; i < 8; i++) {
        int d = lane + 32 * i;
        x[i] = A[ib + d] + Bm[ib + d];
        s += x[i];
    }
    s = wred_sum(s);
    float m = s * (1.f / D_DIM);
    float v = 0.f;
    #pragma unroll
    for (int i = 0; i < 8; i++) { float d = x[i] - m; v += d * d; }
    v = wred_sum(v) * (1.f / D_DIM);
    float inv = rsqrtf(v + EPS);
    size_t ob = (size_t)rOut * D_DIM;
    #pragma unroll
    for (int i = 0; i < 8; i++) {
        int d = lane + 32 * i;
        out[ob + d] = (x[i] - m) * inv * g[d] + be[d];
    }
}

// ---------------------------------------------------------------------------
// Host launcher
// ---------------------------------------------------------------------------
extern "C" void kernel_launch(void* const* d_in, const int* in_sizes, int n_in,
                              void* d_out, int out_size, void* d_ws, size_t ws_size,
                              hipStream_t stream) {
    const float* src        = (const float*)d_in[0];
    const float* in_proj_w  = (const float*)d_in[1];
    const float* in_proj_b  = (const float*)d_in[2];
    const float* out_proj_w = (const float*)d_in[3];
    const float* out_proj_b = (const float*)d_in[4];
    const float* to_qkv_w   = (const float*)d_in[5];
    const float* proj_w     = (const float*)d_in[6];
    const float* proj_b     = (const float*)d_in[7];
    const float* lin1_w     = (const float*)d_in[8];
    const float* lin1_b     = (const float*)d_in[9];
    const float* lin2_w     = (const float*)d_in[10];
    const float* lin2_b     = (const float*)d_in[11];
    const float* n1_g = (const float*)d_in[12];
    const float* n1_b = (const float*)d_in[13];
    const float* n2_g = (const float*)d_in[14];
    const float* n2_b = (const float*)d_in[15];
    const float* n3_g = (const float*)d_in[16];
    const float* n3_b = (const float*)d_in[17];
    const float* cpb_w1 = (const float*)d_in[18];
    const float* cpb_b1 = (const float*)d_in[19];
    const float* cpb_ln_g = (const float*)d_in[20];
    const float* cpb_ln_b = (const float*)d_in[21];
    const float* cpb_w2 = (const float*)d_in[22];
    const float* logit_scale = (const float*)d_in[23];

    char* ws = (char*)d_ws;
    size_t off = 0;
    auto take = [&](size_t bytes) { char* p = ws + off; off += (bytes + 255) & ~(size_t)255; return p; };
    bf16* wb_in   = (bf16*)take(3 * D_DIM * D_DIM * sizeof(bf16));
    bf16* wb_out  = (bf16*)take(D_DIM * D_DIM * sizeof(bf16));
    bf16* wb_qkv2 = (bf16*)take(3 * D_DIM * D_DIM * sizeof(bf16));
    bf16* wb_proj = (bf16*)take(D_DIM * D_DIM * sizeof(bf16));
    bf16* wb_l1   = (bf16*)take((size_t)FF_DIM * D_DIM * sizeof(bf16));
    bf16* wb_l2   = (bf16*)take((size_t)D_DIM * FF_DIM * sizeof(bf16));
    float* pb     = (float*)take(63 * H_DIM * sizeof(float));
    float* scl    = (float*)take(H_DIM * sizeof(float));
    bf16* R0 = (bf16*)take((size_t)ROWS * 3 * D_DIM * sizeof(bf16)); // qkv1 -> qkv2
    char* R1 = take((size_t)ROWS * D_DIM * sizeof(float));           // xp -> po2 -> ffh chunk
    bf16* R2 = (bf16*)take((size_t)ROWS * D_DIM * sizeof(bf16));     // o1 -> o2
    float* R3 = (float*)take((size_t)ROWS * D_DIM * sizeof(float));  // po -> x2n
    char* R4 = take((size_t)ROWS * D_DIM * sizeof(float));           // x1n -> ffo chunk

    float* xp   = (float*)R1;
    bf16*  qkv1 = R0;
    bf16*  o1   = R2;
    float* po   = R3;
    float* x1n  = (float*)R4;
    bf16*  qkv2 = R0;
    bf16*  o2   = R2;
    float* po2  = (float*)R1;
    float* x2n  = R3;
    bf16*  ffh  = (bf16*)R1;   // per-chunk: 10752 x 2048 bf16
    float* ffo  = (float*)R4;  // per-chunk: 10752 x 256 f32
    float* outp = (float*)d_out;

    auto cvt = [&](const float* s, bf16* d, int n) {
        k_f32_to_bf16<<<(n + 255) / 256, 256, 0, stream>>>(s, d, n);
    };
    cvt(in_proj_w,  wb_in,   3 * D_DIM * D_DIM);
    cvt(out_proj_w, wb_out,  D_DIM * D_DIM);
    cvt(to_qkv_w,   wb_qkv2, 3 * D_DIM * D_DIM);
    cvt(proj_w,     wb_proj, D_DIM * D_DIM);
    cvt(lin1_w,     wb_l1,   FF_DIM * D_DIM);
    cvt(lin2_w,     wb_l2,   D_DIM * FF_DIM);

    k_cpb<<<1, 256, 0, stream>>>(cpb_w1, cpb_b1, cpb_ln_g, cpb_ln_b, cpb_w2,
                                 logit_scale, pb, scl);

    // ---- stage 1 ----
    k_pack_x1<<<ROWS, 256, 0, stream>>>(src, xp);
    k_gemm_wmma<false, true, false, true><<<dim3(3 * D_DIM / BN, ROWS / BM), 256, 0, stream>>>(
        xp, wb_in, in_proj_b, qkv1, ROWS, 3 * D_DIM, D_DIM);
    k_attn1<<<TB * H_DIM, 256, 0, stream>>>(qkv1, o1);
    k_gemm_wmma<true, false, false, true><<<dim3(D_DIM / BN, ROWS / BM), 256, 0, stream>>>(
        o1, wb_out, out_proj_b, po, ROWS, D_DIM, D_DIM);
    k_ln<<<ROWS / 8, 256, 0, stream>>>(xp, po, n1_g, n1_b, x1n, 0, ROWS);

    // ---- stage 2 ----
    k_gemm_wmma<false, true, false, false><<<dim3(3 * D_DIM / BN, ROWS / BM), 256, 0, stream>>>(
        x1n, wb_qkv2, nullptr, qkv2, ROWS, 3 * D_DIM, D_DIM);
    k_attn2<<<BJ * H_DIM, 256, 0, stream>>>(qkv2, pb, scl, o2);
    k_gemm_wmma<true, false, false, true><<<dim3(D_DIM / BN, ROWS / BM), 256, 0, stream>>>(
        o2, wb_proj, proj_b, po2, ROWS, D_DIM, D_DIM);
    k_ln<<<ROWS / 8, 256, 0, stream>>>(x1n, po2, n2_g, n2_b, x2n, 1, ROWS);

    // ---- stage 3: FFN, 4 row chunks of 10752 ----
    const int CHROWS = ROWS / 4;  // 10752 = 84 * 128
    for (int c = 0; c < 4; c++) {
        const float* x2nC = x2n + (size_t)c * CHROWS * D_DIM;
        float* outC = outp + (size_t)c * CHROWS * D_DIM;
        k_gemm_wmma<false, true, true, true><<<dim3(FF_DIM / BN, CHROWS / BM), 256, 0, stream>>>(
            x2nC, wb_l1, lin1_b, ffh, CHROWS, FF_DIM, D_DIM);
        k_gemm_wmma<true, false, false, true><<<dim3(D_DIM / BN, CHROWS / BM), 256, 0, stream>>>(
            ffh, wb_l2, lin2_b, ffo, CHROWS, D_DIM, FF_DIM);
        k_ln<<<CHROWS / 8, 256, 0, stream>>>(x2nC, ffo, n3_g, n3_b, outC, 2, CHROWS);
    }
    (void)in_sizes; (void)n_in; (void)out_size; (void)ws_size;
}
